// Model_11879879542411
// MI455X (gfx1250) — matrix operand order repaired and checked by reading, not compiled, not run on hardware
//
#include <hip/hip_runtime.h>

// ---------------- problem constants ----------------
#define B_    128
#define T_    900
#define F_    240
#define H_    400
#define C_    73
#define G_    1600          // 4*H
#define FP    256           // F padded to mult of 32
#define HP    416           // H padded to mult of 32
#define CP    80            // C padded to mult of 16
#define MROWS (B_*T_)       // 115200

typedef __bf16 bf16;
typedef __bf16 bf16x8  __attribute__((ext_vector_type(8)));
typedef __bf16 bf16x16 __attribute__((ext_vector_type(16)));
typedef float  f32x8   __attribute__((ext_vector_type(8)));

__device__ __forceinline__ bf16x16 bcat(bf16x8 lo, bf16x8 hi) {
  return __builtin_shufflevector(lo, hi, 0,1,2,3,4,5,6,7,8,9,10,11,12,13,14,15);
}

__device__ __forceinline__ f32x8 wmma_bf16(bf16x16 a, bf16x16 b, f32x8 c) {
  // (neg_a, A, neg_b, B, c_mod, C, reuse_a, reuse_b)
  return __builtin_amdgcn_wmma_f32_16x16x32_bf16(false, a, false, b, (short)0, c, false, false);
}

__device__ __forceinline__ float sigmoid_f(float x) { return 1.0f / (1.0f + __expf(-x)); }
__device__ __forceinline__ float tanh_f(float x)    { float e = __expf(-2.0f * x); return (1.0f - e) / (1.0f + e); }

// ---------------- staging / conversion kernels ----------------
__global__ void k_conv_x(const float* __restrict__ x, bf16* __restrict__ xb) {
  int idx = blockIdx.x * 256 + threadIdx.x;               // MROWS*FP total
  if (idx >= MROWS * FP) return;
  int row = idx >> 8;                                     // FP == 256
  int k   = idx & 255;
  xb[idx] = (k < F_) ? (bf16)x[(size_t)row * F_ + k] : (bf16)0.0f;
}

__global__ void k_conv_wxT(const float* __restrict__ w, bf16* __restrict__ wt) {
  int idx = blockIdx.x * 256 + threadIdx.x;               // G_*FP total
  if (idx >= G_ * FP) return;
  int g = idx >> 8;
  int k = idx & 255;
  wt[idx] = (k < F_) ? (bf16)w[(size_t)k * G_ + g] : (bf16)0.0f;
}

__global__ void k_conv_whT(const float* __restrict__ w, bf16* __restrict__ wt) {
  int idx = blockIdx.x * 256 + threadIdx.x;               // G_*HP total
  if (idx >= G_ * HP) return;
  int g = idx / HP;
  int k = idx - g * HP;
  wt[idx] = (k < H_) ? (bf16)w[(size_t)k * G_ + g] : (bf16)0.0f;
}

__global__ void k_conv_wdT(const float* __restrict__ w, bf16* __restrict__ wt) {
  int idx = blockIdx.x * 256 + threadIdx.x;               // CP*HP total
  if (idx >= CP * HP) return;
  int n = idx / HP;
  int k = idx - n * HP;
  wt[idx] = (n < C_ && k < H_) ? (bf16)w[(size_t)k * C_ + n] : (bf16)0.0f;
}

// ---------------- GEMM1: xw[t*128+b][G] = x[b,t,:] @ W_x + bias ----------------
// grid (7200, 25), block 32 (1 wave); wave -> one 16x64 strip (A reused 4x), K = 256.
__global__ __launch_bounds__(32)
void k_gemm_xw(const bf16* __restrict__ xb, const bf16* __restrict__ wxT,
               const float* __restrict__ bias, float* __restrict__ xw) {
  const int lane = threadIdx.x & 31;
  const int l15  = lane & 15;
  const int hsel = lane >> 4;
  const int mt   = blockIdx.x;                 // 0..7199
  const int n0   = blockIdx.y * 64;            // strip of 4 n-tiles
  const int row0 = mt * 16;                    // out rows (t*128+b flat)

  // A row for this lane: out_row -> (t,b) -> x row b*T+t
  const int orow = row0 + l15;
  const int t    = orow >> 7;                  // /128
  const int b    = orow & 127;
  const bf16* arow = xb + (size_t)(b * T_ + t) * FP;

  const bf16* brow[4];
  f32x8 acc[4];
#pragma unroll
  for (int j = 0; j < 4; ++j) {
    int col = n0 + j * 16 + l15;
    brow[j] = wxT + (size_t)col * FP;
    float bv = bias[col];
    acc[j] = (f32x8){bv, bv, bv, bv, bv, bv, bv, bv};
  }

#pragma unroll
  for (int k0 = 0; k0 < FP; k0 += 32) {
    bf16x16 a = bcat(*(const bf16x8*)(arow + k0 + 16 * hsel),
                     *(const bf16x8*)(arow + k0 + 16 * hsel + 8));
#pragma unroll
    for (int j = 0; j < 4; ++j) {
      bf16x16 bm = bcat(*(const bf16x8*)(brow[j] + k0 + 16 * hsel),
                        *(const bf16x8*)(brow[j] + k0 + 16 * hsel + 8));
      acc[j] = wmma_bf16(a, bm, acc[j]);
    }
  }

  // xw is streamed exactly once -> non-temporal stores (keep L2 for reused data)
#pragma unroll
  for (int j = 0; j < 4; ++j) {
    float* outp = xw + (size_t)row0 * G_ + n0 + j * 16 + l15;
#pragma unroll
    for (int r = 0; r < 8; ++r)
      __builtin_nontemporal_store(acc[j][r], outp + (size_t)(r + 8 * hsel) * G_);
  }
}

// ---------------- recurrence: 8 blocks x 800 threads (25 waves) ----------------
// Block bg owns batch rows [16*bg, 16*bg+16). Wave w owns j-group w (cols 16w..16w+15
// of each gate) and computes the matching i/f/g/o 16x16 tiles every step.
__global__ __launch_bounds__(800, 1)
void k_lstm(const float* __restrict__ xw, const bf16* __restrict__ whT,
            const int* __restrict__ seqlen, bf16* __restrict__ hs) {
  __shared__ bf16 h_lds[16 * HP];
  __shared__ int  sl[16];

  const int tid = threadIdx.x;
  const int b0  = blockIdx.x * 16;

  for (int i = tid; i < 16 * HP; i += 800) h_lds[i] = (bf16)0.0f;   // h0 = 0 (+ K pad)
  if (tid < 16) sl[tid] = seqlen[b0 + tid];
  __syncthreads();

  const int lane = tid & 31;
  const int wv   = tid >> 5;                   // 0..24  (j-group)
  const int l15  = lane & 15;
  const int hsel = lane >> 4;
  const int jcol = wv * 16 + l15;              // 0..399 within a gate

  const bf16* bI = whT + (size_t)(0 * H_ + jcol) * HP;
  const bf16* bF = whT + (size_t)(1 * H_ + jcol) * HP;
  const bf16* bG = whT + (size_t)(2 * H_ + jcol) * HP;
  const bf16* bO = whT + (size_t)(3 * H_ + jcol) * HP;

  float cst[8];
#pragma unroll
  for (int r = 0; r < 8; ++r) cst[r] = 0.0f;

  const int copy_row   = tid / 50;             // 16 rows x 50 chunks of 8 bf16
  const int copy_chunk = tid - copy_row * 50;

  for (int t = 0; t < T_; ++t) {
    // Issue the xw loads FIRST (non-temporal: streamed once), but consume them
    // only AFTER the matmul -> they overlap the whole 52-WMMA K-loop.
    const float* xwbase = xw + ((size_t)t * B_ + b0) * G_;
    float xi[8], xf[8], xg[8], xo[8];
#pragma unroll
    for (int r = 0; r < 8; ++r) {
      size_t roff = (size_t)(r + 8 * hsel) * G_;
      xi[r] = __builtin_nontemporal_load(xwbase + roff + 0 * H_ + jcol);
      xf[r] = __builtin_nontemporal_load(xwbase + roff + 1 * H_ + jcol);
      xg[r] = __builtin_nontemporal_load(xwbase + roff + 2 * H_ + jcol);
      xo[r] = __builtin_nontemporal_load(xwbase + roff + 3 * H_ + jcol);
    }

    // z = h @ W_h   (A from LDS, shared across the 4 gate tiles)
    f32x8 zi = {}, zf = {}, zg = {}, zo = {};
    for (int k0 = 0; k0 < HP; k0 += 32) {
      bf16x16 a = bcat(*(const bf16x8*)&h_lds[l15 * HP + k0 + 16 * hsel],
                       *(const bf16x8*)&h_lds[l15 * HP + k0 + 16 * hsel + 8]);
      bf16x16 vb;
      vb = bcat(*(const bf16x8*)(bI + k0 + 16 * hsel), *(const bf16x8*)(bI + k0 + 16 * hsel + 8));
      zi = wmma_bf16(a, vb, zi);
      vb = bcat(*(const bf16x8*)(bF + k0 + 16 * hsel), *(const bf16x8*)(bF + k0 + 16 * hsel + 8));
      zf = wmma_bf16(a, vb, zf);
      vb = bcat(*(const bf16x8*)(bG + k0 + 16 * hsel), *(const bf16x8*)(bG + k0 + 16 * hsel + 8));
      zg = wmma_bf16(a, vb, zg);
      vb = bcat(*(const bf16x8*)(bO + k0 + 16 * hsel), *(const bf16x8*)(bO + k0 + 16 * hsel + 8));
      zo = wmma_bf16(a, vb, zo);
    }

    // gate math entirely in registers (i/f/g/o of (row,jcol) all live here)
    float hnew[8];
    bool  msk[8];
#pragma unroll
    for (int r = 0; r < 8; ++r) {
      int row  = r + 8 * hsel;
      bool m   = (t < sl[row]);
      float iv = sigmoid_f(zi[r] + xi[r]);
      float fv = sigmoid_f(zf[r] + xf[r]);
      float gv = tanh_f(zg[r] + xg[r]);
      float ov = sigmoid_f(zo[r] + xo[r]);
      float cn = fv * cst[r] + iv * gv;
      float hn = ov * tanh_f(cn);
      if (m) cst[r] = cn;
      msk[r]  = m;
      hnew[r] = hn;
    }

    __syncthreads();                           // everyone done reading h_lds
#pragma unroll
    for (int r = 0; r < 8; ++r)
      if (msk[r]) h_lds[(r + 8 * hsel) * HP + jcol] = (bf16)hnew[r];
    __syncthreads();                           // h_lds updated

    // stream h (post-mask) to hs[t*128 + b][HP] as bf16 (16B per thread, NT)
    bf16x8 v = *(const bf16x8*)&h_lds[copy_row * HP + copy_chunk * 8];
    __builtin_nontemporal_store(
        v, (bf16x8*)(hs + ((size_t)(t * B_) + b0 + copy_row) * HP + copy_chunk * 8));
  }
}

// ---------------- GEMM3: logits[b][t][c] = hs @ W_d + b_d ----------------
// grid 7200, block 32 (1 wave); wave -> one 16x80 strip (A reused 5x), K = 416.
__global__ __launch_bounds__(32)
void k_gemm_out(const bf16* __restrict__ hs, const bf16* __restrict__ wdT,
                const float* __restrict__ bd, float* __restrict__ out) {
  const int lane = threadIdx.x & 31;
  const int l15  = lane & 15;
  const int hsel = lane >> 4;
  const int mt   = blockIdx.x;
  const int row0 = mt * 16;

  const bf16* arow = hs + (size_t)(row0 + l15) * HP;

  const bf16* brow[5];
  f32x8 acc[5];
#pragma unroll
  for (int j = 0; j < 5; ++j) {
    int col = j * 16 + l15;                    // 0..79
    brow[j] = wdT + (size_t)col * HP;
    float bv = (col < C_) ? bd[col] : 0.0f;
    acc[j] = (f32x8){bv, bv, bv, bv, bv, bv, bv, bv};
  }

#pragma unroll
  for (int k0 = 0; k0 < HP; k0 += 32) {
    bf16x16 a = bcat(*(const bf16x8*)(arow + k0 + 16 * hsel),
                     *(const bf16x8*)(arow + k0 + 16 * hsel + 8));
#pragma unroll
    for (int j = 0; j < 5; ++j) {
      bf16x16 bm = bcat(*(const bf16x8*)(brow[j] + k0 + 16 * hsel),
                        *(const bf16x8*)(brow[j] + k0 + 16 * hsel + 8));
      acc[j] = wmma_bf16(a, bm, acc[j]);
    }
  }

#pragma unroll
  for (int j = 0; j < 5; ++j) {
    int col = j * 16 + l15;
    if (col < C_) {
#pragma unroll
      for (int r = 0; r < 8; ++r) {
        int orow = row0 + r + 8 * hsel;        // t*128 + b
        int t = orow >> 7;
        int b = orow & 127;
        __builtin_nontemporal_store(acc[j][r], out + ((size_t)b * T_ + t) * C_ + col);
      }
    }
  }
}

// ---------------- host launcher ----------------
extern "C" void kernel_launch(void* const* d_in, const int* in_sizes, int n_in,
                              void* d_out, int out_size, void* d_ws, size_t ws_size,
                              hipStream_t stream) {
  (void)in_sizes; (void)n_in; (void)out_size; (void)ws_size;
  const float* x      = (const float*)d_in[0];
  const float* W_x    = (const float*)d_in[1];
  const float* W_h    = (const float*)d_in[2];
  const float* bvec   = (const float*)d_in[3];
  const float* W_d    = (const float*)d_in[4];
  const float* b_d    = (const float*)d_in[5];
  const int*   seqlen = (const int*)d_in[6];
  float*       out    = (float*)d_out;

  char* ws = (char*)d_ws;
  bf16*  xb  = (bf16*)ws;  ws += (size_t)MROWS * FP * sizeof(bf16);   // 58.98 MB
  bf16*  wxT = (bf16*)ws;  ws += (size_t)G_ * FP * sizeof(bf16);      // 0.82 MB
  bf16*  whT = (bf16*)ws;  ws += (size_t)G_ * HP * sizeof(bf16);      // 1.33 MB
  bf16*  wdT = (bf16*)ws;  ws += (size_t)CP * HP * sizeof(bf16);      // 0.07 MB
  float* xw  = (float*)ws; ws += (size_t)MROWS * G_ * sizeof(float);  // 737 MB
  bf16*  hsb = (bf16*)ws;  ws += (size_t)MROWS * HP * sizeof(bf16);   // 95.8 MB

  k_conv_x  <<<(MROWS * FP) / 256, 256, 0, stream>>>(x, xb);
  k_conv_wxT<<<(G_ * FP) / 256,    256, 0, stream>>>(W_x, wxT);
  k_conv_whT<<<(G_ * HP) / 256,    256, 0, stream>>>(W_h, whT);
  k_conv_wdT<<<(CP * HP) / 256,    256, 0, stream>>>(W_d, wdT);

  k_gemm_xw <<<dim3(MROWS / 16, 25), 32, 0, stream>>>(xb, wxT, bvec, xw);
  k_lstm    <<<B_ / 16, 800, 0, stream>>>(xw, whT, seqlen, hsb);
  k_gemm_out<<<MROWS / 16, 32, 0, stream>>>(hsb, wdT, b_d, out);
}
